// NattenAttention_85109071938185
// MI455X (gfx1250) — matrix operand order repaired and checked by reading, not compiled, not run on hardware
//
#include <hip/hip_runtime.h>
#include <cstdint>
#include <cstddef>

// ---------------------------------------------------------------------------
// NATTEN-style neighborhood attention for MI455X (gfx1250, wave32, WMMA).
//   B=2, C=128, H=W=96, HEADS=8, hd=16, K=8 (64-neighbor window), GROUPS=32.
// GEMMs (QKV 384x128, proj 128x128 over 18432 pixels) run on
// v_wmma_f32_16x16x32_f16 (f16 in, f32 accumulate). Activation tiles are
// staged to LDS via the gfx1250 async global->LDS path (ASYNCcnt) and fed to
// WMMA from ds_load_b128. GroupNorm, softmax and the windowed attention run
// in fp32 on VALU with float4 (b128) loads.
// ---------------------------------------------------------------------------

#define BATCH   2
#define CCH     128          // channels
#define HDIM    96
#define WDIM    96
#define HW      (HDIM * WDIM)        // 9216
#define NPIX    (BATCH * HW)         // 18432
#define NHEADS  8
#define HD      16
#define KWIN    8                    // neighborhood size (8x8 = 64)
#define GROUPS  32
#define CPG     (CCH / GROUPS)       // 4
#define GN_N    (CPG * HW)           // 36864 elements per (b, group)

typedef __attribute__((ext_vector_type(16))) _Float16 v16h;
typedef __attribute__((ext_vector_type(8)))  float    v8f;
typedef int v4i __attribute__((vector_size(16)));

union Frag {
    v16h     h;
    uint32_t u[8];
};

#if defined(__AMDGCN__) && __has_builtin(__builtin_amdgcn_global_load_async_to_lds_b128)
#define HAVE_ASYNC_LDS 1
typedef __attribute__((address_space(1))) v4i g_v4i;   // global int4
typedef __attribute__((address_space(3))) v4i l_v4i;   // LDS int4
#else
#define HAVE_ASYNC_LDS 0
#endif

// ---------------------------------------------------------------------------
// 1) GroupNorm statistics: one block per (b, group). Channels of a group are
//    contiguous in x (B, C, H, W), so each (b,g) chunk is 4*9216 floats.
// ---------------------------------------------------------------------------
__global__ void gn_stats_kernel(const float* __restrict__ x,
                                float* __restrict__ stats /* [64][2] */) {
    const int bg = blockIdx.x;                       // b*32 + g
    const float* p = x + (size_t)bg * GN_N;
    float s = 0.f, ss = 0.f;
    for (int i = threadIdx.x; i < GN_N; i += 256) {
        float v = p[i];
        s  += v;
        ss += v * v;
    }
    __shared__ float sh0[256];
    __shared__ float sh1[256];
    sh0[threadIdx.x] = s;
    sh1[threadIdx.x] = ss;
    __syncthreads();
    for (int o = 128; o > 0; o >>= 1) {
        if (threadIdx.x < o) {
            sh0[threadIdx.x] += sh0[threadIdx.x + o];
            sh1[threadIdx.x] += sh1[threadIdx.x + o];
        }
        __syncthreads();
    }
    if (threadIdx.x == 0) {
        const float inv_n = 1.0f / (float)GN_N;
        float mean = sh0[0] * inv_n;
        float var  = sh1[0] * inv_n - mean * mean;
        stats[bg * 2 + 0] = mean;
        stats[bg * 2 + 1] = rsqrtf(var + 1e-6f);
    }
}

// ---------------------------------------------------------------------------
// 2) Apply GN, cast to f16, transpose to pixel-major h_t[(b*HW+p)*128 + c]
//    so WMMA B-fragments and attention reads are K-contiguous.
// ---------------------------------------------------------------------------
__global__ void gn_apply_kernel(const float* __restrict__ x,
                                const float* __restrict__ stats,
                                const float* __restrict__ gamma,
                                const float* __restrict__ beta,
                                _Float16* __restrict__ h_t) {
    const size_t id = (size_t)blockIdx.x * 256 + threadIdx.x;  // b,c,p order
    const int p = (int)(id % HW);
    const size_t t = id / HW;
    const int c = (int)(t % CCH);
    const int b = (int)(t / CCH);
    const int bg = b * GROUPS + (c >> 2);
    const float mean = stats[2 * bg + 0];
    const float rstd = stats[2 * bg + 1];
    const float v = (x[id] - mean) * rstd * gamma[c] + beta[c];
    h_t[((size_t)b * HW + p) * CCH + c] = (_Float16)v;
}

// ---------------------------------------------------------------------------
// 3) Weight fp32 -> f16 cast (qkv_w 384x128, proj_w 128x128, row-major).
// ---------------------------------------------------------------------------
__global__ void cvt_weights_kernel(const float* __restrict__ qkv_w,
                                   const float* __restrict__ proj_w,
                                   _Float16* __restrict__ wq,
                                   _Float16* __restrict__ wp) {
    const int id = blockIdx.x * 256 + threadIdx.x;
    const int NQ = 384 * 128;
    const int NP = 128 * 128;
    if (id < NQ) {
        wq[id] = (_Float16)qkv_w[id];
    } else if (id < NQ + NP) {
        wp[id - NQ] = (_Float16)proj_w[id - NQ];
    }
}

// ---------------------------------------------------------------------------
// 4) WMMA GEMM with LDS-staged activations:
//      out[m, n] = sum_k A[m,k] * Bt[n,k] + bias[m]
//    A:  M x 128 row-major f16 (weights, L2-resident)
//    Bt: Ntot x 128 pixel-major f16 (activations, K contiguous per pixel)
//    Block = 8 waves = 8 M-tiles x one 64-column N super-tile. The 16 KB
//    contiguous B super-tile is staged to LDS once per block with
//    global_load_async_to_lds_b128 (ASYNCcnt), then every wave runs the
//    K-loop (4 steps) -> 16 v_wmma per wave, B-fragments from ds_load.
//    mode 0: out[n*M + m]            (pixel-major, QKV output)
//    mode 1: out[b*M*HW + m*HW + p]  (channel-major, final projection)
// ---------------------------------------------------------------------------
__global__ void gemm_wmma_kernel(const _Float16* __restrict__ A,
                                 const _Float16* __restrict__ Bt,
                                 const float* __restrict__ bias,
                                 float* __restrict__ out,
                                 int M, int mgroups, int mode) {
    __shared__ _Float16 btile[64 * 128];             // 16 KB: 64 pixels x 128 K

    const int mgroup = blockIdx.x % mgroups;
    const int ns     = blockIdx.x / mgroups;

    // ---- Stage B super-tile (contiguous 16 KB) into LDS -------------------
    {
        const _Float16* bsrc = Bt + (size_t)ns * 64 * 128;
#if HAVE_ASYNC_LDS
        #pragma unroll
        for (int r = 0; r < 4; ++r) {
            const int chunk = threadIdx.x + r * 256;     // 1024 x 16B chunks
            const _Float16* gp = bsrc + chunk * 8;
            _Float16* lp = &btile[chunk * 8];
            __builtin_amdgcn_global_load_async_to_lds_b128(
                (g_v4i*)(v4i*)gp,
                (l_v4i*)(v4i*)lp,
                0, 0);
        }
#if __has_builtin(__builtin_amdgcn_s_wait_asynccnt)
        __builtin_amdgcn_s_wait_asynccnt(0);
#else
        asm volatile("s_wait_asynccnt 0x0" ::: "memory");
#endif
#else
        #pragma unroll
        for (int r = 0; r < 4; ++r) {
            const int chunk = threadIdx.x + r * 256;
            *(float4*)&btile[chunk * 8] = *(const float4*)(bsrc + chunk * 8);
        }
#endif
    }
    __syncthreads();

    // ---- Per-wave 16(M) x 64(N) strip ------------------------------------
    const int wv   = (int)(threadIdx.x >> 5);        // wave in block: 0..7
    const int lane = (int)(threadIdx.x & 31);
    const int hl   = lane >> 4;                      // half: lanes 0-15 / 16-31
    const int l15  = lane & 15;
    const int m0   = (mgroup * 8 + wv) << 4;

    v8f acc0 = {}, acc1 = {}, acc2 = {}, acc3 = {};
    const _Float16* arow = A + (size_t)(m0 + l15) * 128;

    #pragma unroll
    for (int kt = 0; kt < 4; ++kt) {
        const int kb = kt << 5;                      // 32 K per step
        // A fragment: 16x32 f16, ISA layout:
        //   VGPR v holds K pair at (v>=4?16:0) + half*8 + (v&3)*2
        Frag a;
        #pragma unroll
        for (int v = 0; v < 8; ++v) {
            const int k = kb + ((v >> 2) << 4) + (hl << 3) + ((v & 3) << 1);
            a.u[v] = *(const uint32_t*)(arow + k);
        }
        // 4 B fragments from LDS: column n = lane&15,
        //   VGPR v holds K pair at (v>=4?16:0) + half*8 + (v&3)*2
        #pragma unroll
        for (int t = 0; t < 4; ++t) {
            const _Float16* brow = &btile[(t * 16 + l15) * 128 + kb + (hl << 3)];
            Frag bf;
            #pragma unroll
            for (int v = 0; v < 8; ++v) {
                bf.u[v] = *(const uint32_t*)(brow + ((v >> 2) << 4) + ((v & 3) << 1));
            }
            v8f* accp = (t == 0) ? &acc0 : (t == 1) ? &acc1 : (t == 2) ? &acc2 : &acc3;
            *accp = __builtin_amdgcn_wmma_f32_16x16x32_f16(
                false, a.h, false, bf.h, (short)0, *accp, false, false);
        }
    }

    // Epilogue: lane holds rows m = m0 + half*8 + r (r = 0..7), col n fixed.
    float bv[8];
    #pragma unroll
    for (int r = 0; r < 8; ++r) bv[r] = bias[m0 + (hl << 3) + r];

    #pragma unroll
    for (int t = 0; t < 4; ++t) {
        const v8f acc = (t == 0) ? acc0 : (t == 1) ? acc1 : (t == 2) ? acc2 : acc3;
        const int n = (ns << 6) + (t << 4) + l15;
        if (mode == 0) {
            float4 lo, hi;
            lo.x = acc[0] + bv[0]; lo.y = acc[1] + bv[1];
            lo.z = acc[2] + bv[2]; lo.w = acc[3] + bv[3];
            hi.x = acc[4] + bv[4]; hi.y = acc[5] + bv[5];
            hi.z = acc[6] + bv[6]; hi.w = acc[7] + bv[7];
            float* dst = out + (size_t)n * M + m0 + (hl << 3);
            *(float4*)(dst)     = lo;
            *(float4*)(dst + 4) = hi;
        } else {
            const int b   = n / HW;
            const int pix = n % HW;
            float* dst = out + (size_t)b * M * HW + pix;
            #pragma unroll
            for (int r = 0; r < 8; ++r) {
                dst[(size_t)(m0 + (hl << 3) + r) * HW] = acc[r] + bv[r];
            }
        }
    }
}

// ---------------------------------------------------------------------------
// 5) Neighborhood attention, one thread per (pixel, head). Online softmax
//    over the 64 neighbors; K/V vectors read as contiguous float4s from the
//    pixel-major qkv tensor. Output cast to f16 pixel-major for proj GEMM.
// ---------------------------------------------------------------------------
__global__ void natten_kernel(const float* __restrict__ qkv_t /* [NPIX][384] */,
                              _Float16* __restrict__ attn_t   /* [NPIX][128] */) {
    const int g = blockIdx.x * 256 + threadIdx.x;   // [0, NPIX*NHEADS)
    const int head = g & 7;
    const int n    = g >> 3;                        // global pixel
    const int b    = n / HW;
    const int pix  = n % HW;
    const int i    = pix / WDIM;                    // row (H)
    const int j    = pix % WDIM;                    // col (W)

    // window start: clip(i - 3, 0, L - 8)
    int si = i - 3; si = si < 0 ? 0 : si; si = si > (HDIM - KWIN) ? (HDIM - KWIN) : si;
    int sj = j - 3; sj = sj < 0 ? 0 : sj; sj = sj > (WDIM - KWIN) ? (WDIM - KWIN) : sj;

    const float scale = 0.25f;                      // 1/sqrt(16)
    const float* qp = qkv_t + (size_t)n * 384 + head * HD;
    float4 q0 = *(const float4*)(qp + 0);
    float4 q1 = *(const float4*)(qp + 4);
    float4 q2 = *(const float4*)(qp + 8);
    float4 q3 = *(const float4*)(qp + 12);
    q0.x *= scale; q0.y *= scale; q0.z *= scale; q0.w *= scale;
    q1.x *= scale; q1.y *= scale; q1.z *= scale; q1.w *= scale;
    q2.x *= scale; q2.y *= scale; q2.z *= scale; q2.w *= scale;
    q3.x *= scale; q3.y *= scale; q3.z *= scale; q3.w *= scale;

    float acc[HD];
    #pragma unroll
    for (int c = 0; c < HD; ++c) acc[c] = 0.f;
    float mrun = -1e30f, srun = 0.f;

    for (int wi = 0; wi < KWIN; ++wi) {
        const int ni = si + wi;
        for (int wj = 0; wj < KWIN; ++wj) {
            const int np = ni * WDIM + (sj + wj);
            const float* kp = qkv_t + ((size_t)b * HW + np) * 384 + CCH + head * HD;
            const float* vp = kp + CCH;
            float4 k0 = *(const float4*)(kp + 0);
            float4 k1 = *(const float4*)(kp + 4);
            float4 k2 = *(const float4*)(kp + 8);
            float4 k3 = *(const float4*)(kp + 12);
            float dot =
                q0.x * k0.x + q0.y * k0.y + q0.z * k0.z + q0.w * k0.w +
                q1.x * k1.x + q1.y * k1.y + q1.z * k1.z + q1.w * k1.w +
                q2.x * k2.x + q2.y * k2.y + q2.z * k2.z + q2.w * k2.w +
                q3.x * k3.x + q3.y * k3.y + q3.z * k3.z + q3.w * k3.w;
            const float mnew = dot > mrun ? dot : mrun;
            const float corr = __expf(mrun - mnew);
            const float e    = __expf(dot - mnew);
            srun = srun * corr + e;
            float4 v0 = *(const float4*)(vp + 0);
            float4 v1 = *(const float4*)(vp + 4);
            float4 v2 = *(const float4*)(vp + 8);
            float4 v3 = *(const float4*)(vp + 12);
            acc[0]  = acc[0]  * corr + e * v0.x;
            acc[1]  = acc[1]  * corr + e * v0.y;
            acc[2]  = acc[2]  * corr + e * v0.z;
            acc[3]  = acc[3]  * corr + e * v0.w;
            acc[4]  = acc[4]  * corr + e * v1.x;
            acc[5]  = acc[5]  * corr + e * v1.y;
            acc[6]  = acc[6]  * corr + e * v1.z;
            acc[7]  = acc[7]  * corr + e * v1.w;
            acc[8]  = acc[8]  * corr + e * v2.x;
            acc[9]  = acc[9]  * corr + e * v2.y;
            acc[10] = acc[10] * corr + e * v2.z;
            acc[11] = acc[11] * corr + e * v2.w;
            acc[12] = acc[12] * corr + e * v3.x;
            acc[13] = acc[13] * corr + e * v3.y;
            acc[14] = acc[14] * corr + e * v3.z;
            acc[15] = acc[15] * corr + e * v3.w;
            mrun = mnew;
        }
    }
    const float inv = 1.0f / srun;
    _Float16* op = attn_t + (size_t)n * CCH + head * HD;
    #pragma unroll
    for (int c = 0; c < HD; ++c) op[c] = (_Float16)(acc[c] * inv);
}

// ---------------------------------------------------------------------------
// Launch: inputs (all fp32) in setup_inputs() order:
//   0:x  1:gn_gamma  2:gn_beta  3:qkv_w  4:qkv_b  5:proj_w  6:proj_b
// ---------------------------------------------------------------------------
extern "C" void kernel_launch(void* const* d_in, const int* in_sizes, int n_in,
                              void* d_out, int out_size, void* d_ws, size_t ws_size,
                              hipStream_t stream) {
    const float* x      = (const float*)d_in[0];
    const float* gamma  = (const float*)d_in[1];
    const float* beta   = (const float*)d_in[2];
    const float* qkv_w  = (const float*)d_in[3];
    const float* qkv_b  = (const float*)d_in[4];
    const float* proj_w = (const float*)d_in[5];
    const float* proj_b = (const float*)d_in[6];
    float* out = (float*)d_out;

    // Workspace carve-up
    char* w = (char*)d_ws;
    const size_t SZ_HT   = (size_t)NPIX * CCH * sizeof(_Float16);   //  4,718,592
    const size_t SZ_QKV  = (size_t)NPIX * 384 * sizeof(float);      // 28,311,552
    const size_t SZ_ATT  = (size_t)NPIX * CCH * sizeof(_Float16);   //  4,718,592
    const size_t SZ_WQ   = (size_t)384 * 128 * sizeof(_Float16);    //     98,304
    const size_t SZ_WP   = (size_t)128 * 128 * sizeof(_Float16);    //     32,768
    size_t off = 0;
    _Float16* h_t    = (_Float16*)(w + off); off += SZ_HT;
    float*    qkv_t  = (float*)   (w + off); off += SZ_QKV;
    _Float16* attn_t = (_Float16*)(w + off); off += SZ_ATT;
    _Float16* wq     = (_Float16*)(w + off); off += SZ_WQ;
    _Float16* wp     = (_Float16*)(w + off); off += SZ_WP;
    float*    stats  = (float*)   (w + off); off += 64 * 2 * sizeof(float);
    (void)ws_size; (void)in_sizes; (void)n_in; (void)out_size;

    // 1) GroupNorm statistics: 64 (b,group) blocks
    gn_stats_kernel<<<BATCH * GROUPS, 256, 0, stream>>>(x, stats);

    // 2) Normalize + cast + transpose to pixel-major f16
    gn_apply_kernel<<<(BATCH * CCH * HW) / 256, 256, 0, stream>>>(
        x, stats, gamma, beta, h_t);

    // 3) Weight casts
    cvt_weights_kernel<<<(384 * 128 + 128 * 128) / 256, 256, 0, stream>>>(
        qkv_w, proj_w, wq, wp);

    // 4) QKV GEMM: M=384 -> 3 M-groups x 288 N-supers = 864 blocks
    gemm_wmma_kernel<<<3 * (NPIX / 64), 256, 0, stream>>>(
        wq, h_t, qkv_b, qkv_t, 384, /*mgroups=*/3, /*mode=*/0);

    // 5) Neighborhood attention: one thread per (pixel, head)
    natten_kernel<<<(NPIX * NHEADS) / 256, 256, 0, stream>>>(qkv_t, attn_t);

    // 6) Output projection: M=128 -> 1 M-group x 288 N-supers = 288 blocks
    gemm_wmma_kernel<<<1 * (NPIX / 64), 256, 0, stream>>>(
        wp, attn_t, proj_b, out, 128, /*mgroups=*/1, /*mode=*/1);
}
